// CapsNet_80169859547221
// MI455X (gfx1250) — matrix operand order repaired and checked by reading, not compiled, not run on hardware
//
#include <hip/hip_runtime.h>
#include <math.h>

// Problem constants
#define B_   64
#define I_   4096
#define K8   8
#define O_   32
#define J_   16

typedef __attribute__((ext_vector_type(16))) _Float16 v16h;
typedef __attribute__((ext_vector_type(8)))  _Float16 v8h;
typedef __attribute__((ext_vector_type(8)))  float    v8f;

// ---------------------------------------------------------------------------
// WMMA 16x16x32 f16 lane layouts (CDNA5 ISA §7.12.2), wave32:
//  A (16x32, M x K): lane L holds row M=L%16.  halves h:
//      K = h + (h>=8 ? 8 : 0) + (L>=16 ? 8 : 0)
//  B (32x16, K x N): lane L holds col N=L%16.  halves h:  K = h + (h>=8 ? 8 : 0) + (L>=16 ? 8 : 0)
//  C/D (16x16 f32): lane L holds col N=L%16, vgpr r -> row M = r + 8*(L/16)
// ---------------------------------------------------------------------------

// Pack W (fp32 [i][o][j][k]) into fp16 B-operand tiles for the s-GEMM:
// Wp[o][kt][lane][h], tile kt covers kk=kt*32..+31 where kk=i*8+k, N=j.
__global__ void pack_wp(const float* __restrict__ W, _Float16* __restrict__ Wp) {
  int id = blockIdx.x * 256 + threadIdx.x;       // 16,777,216 total, exact
  int h  = id & 15;
  int L  = (id >> 4) & 31;
  int kt = (id >> 9) & 1023;
  int o  = id >> 19;
  int K  = h + ((h >= 8) ? 8 : 0) + ((L >> 4) << 3); // B layout K
  int j  = L & 15;                               // B layout N
  int kk = kt * 32 + K;
  int i  = kk >> 3, k = kk & 7;
  Wp[id] = (_Float16)W[((i * O_ + o) * J_ + j) * K8 + k];
}

// Pack W as the A-operand of the agreement GEMM:
// rows M=(di,k) (2 in-capsules x 8 dims), K-dim=j (16 real, upper 8 halves of
// each lane are zero-pad added at load time).  Wq[o][ip][lane][h(0..7)]:
//   M = lane%16 -> di=M/8, k=M%8;  j = h + 8*(lane/16);  i = ip*2+di.
__global__ void pack_wq(const float* __restrict__ W, _Float16* __restrict__ Wq) {
  int id = blockIdx.x * 256 + threadIdx.x;       // 16,777,216 total, exact
  int h  = id & 7;
  int L  = (id >> 3) & 31;
  int ip = (id >> 8) & 2047;
  int o  = id >> 19;
  int M  = L & 15;
  int di = M >> 3, k = M & 7;
  int j  = h + ((L >> 4) << 3);
  int i  = ip * 2 + di;
  Wq[id] = (_Float16)W[((i * O_ + o) * J_ + j) * K8 + k];
}

// Pack u (fp32 [b][i][k]) into fp16 A-operand tiles: Up[b4][kt][lane][h].
// Also zero-fills the 16-half block Zh (passed to caps_r, which pads with literal 0s).
__global__ void pack_u(const float* __restrict__ u, _Float16* __restrict__ Up,
                       _Float16* __restrict__ Zh) {
  int id = blockIdx.x * 256 + threadIdx.x;       // 2,097,152 total, exact
  int h  = id & 15;
  int L  = (id >> 4) & 31;
  int kt = (id >> 9) & 1023;
  int b4 = id >> 19;
  int b  = b4 * 16 + (L & 15);                   // A layout M
  int K  = h + ((h >= 8) ? 8 : 0) + ((L >= 16) ? 8 : 0);
  int kk = kt * 32 + K;
  int i  = kk >> 3, k = kk & 7;
  Up[id] = (_Float16)u[(b * I_ + i) * K8 + k];
  if (id < 16) Zh[id] = (_Float16)0.0f;
}

// s-GEMM: spart[kc][b][o][j] = sum over K-chunk kc of (c*u) x Wp.
// One wave per (o, b-tile, k-chunk).  ROUTED=false: uniform c folded into
// the squash kernel's scale, so A is just the pre-packed fp16 u.
template <bool ROUTED>
__global__ __launch_bounds__(32)
void caps_s(const _Float16* __restrict__ Up, const _Float16* __restrict__ Wp,
            const float* __restrict__ c, float* __restrict__ spart) {
  const int o  = blockIdx.x;           // 32
  const int b4 = blockIdx.y;           // 4 b-tiles
  const int tb = b4 * 16;
  const int kc = blockIdx.z;           // 16 K-chunks
  const int L  = threadIdx.x;
  const int bb = tb + (L & 15);        // A-operand row (batch)
  const int off = (L >= 16) ? 8 : 0;   // A-layout K offset for upper lanes
  const _Float16* wpo = Wp + (size_t)o * 524288;
  const _Float16* upo = Up + (size_t)b4 * 524288;

  v8f acc;
  #pragma unroll
  for (int r = 0; r < 8; ++r) acc[r] = 0.0f;

  for (int s = 0; s < 64; ++s) {       // 64 K-steps of 32 -> K=2048 per chunk
    int kt = kc * 64 + s;
    v16h a = *(const v16h*)(upo + (kt * 32 + L) * 16);
    if (ROUTED) {
      int ia = ((kt * 32) + off) >> 3;           // in-capsule idx, this lane
      _Float16 hA = (_Float16)c[(bb * I_ + ia) * O_ + o];
      _Float16 hB = (_Float16)c[(bb * I_ + ia + 2) * O_ + o];
      #pragma unroll
      for (int h = 0; h < 8; ++h) a[h]     = a[h]     * hA;  // v_pk_mul_f16
      #pragma unroll
      for (int h = 0; h < 8; ++h) a[h + 8] = a[h + 8] * hB;
    }
    v16h bm = *(const v16h*)(wpo + (kt * 32 + L) * 16);
    __builtin_prefetch(wpo + ((kt + 1) * 32 + L) * 16, 0, 3);

    acc = __builtin_amdgcn_wmma_f32_16x16x32_f16(false, a, false, bm,
                                                 (short)0, acc, false, false);
  }
  const int n = L & 15;
  #pragma unroll
  for (int r = 0; r < 8; ++r) {
    int m = r + ((L >> 4) << 3);
    spart[((kc * B_ + tb + m) * O_ + o) * J_ + n] = acc[r];
  }
}

// Reduce K-chunk partials, apply coupling scale, squash.  Writes v as fp16
// (B-operand feed for caps_r) and, on the final iteration, fp32 to d_out.
__global__ void caps_v(const float* __restrict__ spart, _Float16* __restrict__ vh,
                       float* __restrict__ out, float scale) {
  int t = blockIdx.x * 256 + threadIdx.x;        // 2048 = B_*O_
  if (t >= B_ * O_) return;
  int b = t / O_, o = t % O_;
  float s[16];
  #pragma unroll
  for (int j = 0; j < 16; ++j) s[j] = 0.0f;
  for (int kc = 0; kc < 16; ++kc) {
    const float* p = spart + ((kc * B_ + b) * O_ + o) * J_;
    #pragma unroll
    for (int j = 0; j < 16; ++j) s[j] += p[j];
  }
  float sq = 0.0f;
  #pragma unroll
  for (int j = 0; j < 16; ++j) { s[j] *= scale; sq += s[j] * s[j]; }
  float sc = sq / ((1.0f + sq) * sqrtf(sq + 1e-8f));
  #pragma unroll
  for (int j = 0; j < 16; ++j) {
    float vj = s[j] * sc;
    vh[(b * O_ + o) * J_ + j] = (_Float16)vj;
    if (out) out[(b * O_ + o) * J_ + j] = vj;
  }
}

// Agreement + softmax, LDS-free and register-lean (rolled o-loop):
//   G'[(di,k), b] = sum_j W[i,o,j,k] * v[b,o,j]   (WMMA: A=W, B=v, K=j pad32)
// D layout puts k on the accumulator-register axis, so
//   bu[b,i,o] = sum_r g[r] * u[b,i,r]             (in-lane 8-term FMA)
// and each lane owns exactly one (b,i).  Updated logits stream to blog[o]
// (same-thread store->load is ordered, ISA §7.3); softmax runs as two more
// rolled sweeps so no 32-float register array is needed.
template <bool FIRST>
__global__ __launch_bounds__(32)
void caps_r(const float* __restrict__ u, const _Float16* __restrict__ Wq,
            const _Float16* __restrict__ vh, const _Float16* __restrict__ Zh,
            float* __restrict__ blog, float* __restrict__ c) {
  const int ip = blockIdx.x;                     // 2048 i-pairs
  const int tb = blockIdx.y * 16;                // 4 b-tiles
  const int L  = threadIdx.x;
  const int b  = tb + (L & 15);                  // D column owned by this lane
  const int di = L >> 4;                         // D row half -> in-capsule
  const int i  = ip * 2 + di;

  // u[b,i,0..7]: contiguous, hoisted out of the o-loop
  const float* pu = u + (b * I_ + i) * K8;
  float ur[8];
  #pragma unroll
  for (int r = 0; r < 8; ++r) ur[r] = pu[r];

  const _Float16* wq0 = Wq + (size_t)ip * 256 + L * 8;
  // lane half walks v[b,o,8*(L/16)..+7]; upper 8 halves zero-padded
  const _Float16* pb0 = vh + (size_t)(b * O_) * J_ + ((L >> 4) << 3);
  const int pbstride  = J_;

  float* bl = blog + (b * I_ + i) * O_;
  float* cp = c    + (b * I_ + i) * O_;

  float mx = -1e30f;
  for (int o = 0; o < O_; ++o) {                 // rolled: ~50 VGPRs live
    v8h w8 = *(const v8h*)(wq0 + (size_t)o * 524288);
    v16h a;
    #pragma unroll
    for (int h = 0; h < 8; ++h)  a[h]     = w8[h];
    #pragma unroll
    for (int h = 0; h < 8; ++h)  a[h + 8] = (_Float16)0.0f;
    v8h v8 = *(const v8h*)(pb0 + o * pbstride);
    v16h bv;
    #pragma unroll
    for (int h = 0; h < 8; ++h)  bv[h]     = v8[h];
    #pragma unroll
    for (int h = 0; h < 8; ++h)  bv[h + 8] = (_Float16)0.0f;

    v8f g;
    #pragma unroll
    for (int r = 0; r < 8; ++r) g[r] = 0.0f;
    g = __builtin_amdgcn_wmma_f32_16x16x32_f16(false, a, false, bv,
                                               (short)0, g, false, false);
    float t = 0.0f;
    #pragma unroll
    for (int r = 0; r < 8; ++r) t += g[r] * ur[r];  // k-dot, in-lane
    if (!FIRST) t += bl[o];
    bl[o] = t;
    mx = fmaxf(mx, t);
  }

  float sum = 0.0f;
  for (int o = 0; o < O_; ++o) {                 // rolled re-read (L2-hot)
    float e = __expf(bl[o] - mx);
    sum += e;
    cp[o] = e;                                   // unnormalized
  }
  float inv = 1.0f / sum;
  for (int o = 0; o < O_; ++o) cp[o] *= inv;     // normalize in place
}

extern "C" void kernel_launch(void* const* d_in, const int* in_sizes, int n_in,
                              void* d_out, int out_size, void* d_ws, size_t ws_size,
                              hipStream_t stream) {
  const float* u = (const float*)d_in[0];        // [64,4096,8]
  const float* W = (const float*)d_in[1];        // [4096,32,16,8]
  float* out = (float*)d_out;                    // [64,32,16]

  char* ws = (char*)d_ws;
  _Float16* Wp    = (_Float16*)(ws);                       // 33.5 MB
  _Float16* Wq    = (_Float16*)(ws + 33554432);            // 33.5 MB
  float*    cbuf  = (float*)   (ws + 67108864);            // 33.5 MB
  float*    blog  = (float*)   (ws + 100663296);           // 33.5 MB
  float*    spart = (float*)   (ws + 134217728);           //  2.0 MB
  _Float16* vh    = (_Float16*)(ws + 136314880);           //  64 KB
  _Float16* Zh    = (_Float16*)(ws + 136380416);           //  32 B
  _Float16* Up    = (_Float16*)(ws + 136380448);           //  4.0 MB

  // One-time fp32 -> fp16 packing of W and u into WMMA-native tile layouts
  pack_wp<<<65536, 256, 0, stream>>>(W, Wp);
  pack_wq<<<65536, 256, 0, stream>>>(W, Wq);
  pack_u <<< 8192, 256, 0, stream>>>(u, Up, Zh);

  dim3 gS(32, 4, 16);     // (o, b-tile, k-chunk)
  dim3 gR(2048, 4);       // (i-pair, b-tile)

  // Iter 1: c uniform (softmax of zeros) -> 1/32 folded into squash scale
  caps_s<false><<<gS, 32, 0, stream>>>(Up, Wp, nullptr, spart);
  caps_v<<<8, 256, 0, stream>>>(spart, vh, nullptr, 0.03125f);
  // Iter 2
  caps_r<true><<<gR, 32, 0, stream>>>(u, Wq, vh, Zh, blog, cbuf);
  caps_s<true><<<gS, 32, 0, stream>>>(Up, Wp, cbuf, spart);
  caps_v<<<8, 256, 0, stream>>>(spart, vh, nullptr, 1.0f);
  // Iter 3 (its trailing b-update is dead in the reference -> skipped)
  caps_r<false><<<gR, 32, 0, stream>>>(u, Wq, vh, Zh, blog, cbuf);
  caps_s<true><<<gS, 32, 0, stream>>>(Up, Wp, cbuf, spart);
  caps_v<<<8, 256, 0, stream>>>(spart, vh, out, 1.0f);
}
